// H2GCN_78176994721832
// MI455X (gfx1250) — matrix operand order repaired and checked by reading, not compiled, not run on hardware
//
#include <hip/hip_runtime.h>
#include <hip/hip_bf16.h>

typedef __attribute__((ext_vector_type(16))) _Float16 v16h;
typedef __attribute__((ext_vector_type(8)))  float    v8f;

#define IN_DIM 500
#define HID 64
#define NCLS 47
#define USE_ASYNC 1

// ---------------------------------------------------------------------------
// Pack fp32 weight matrix W[Kin, Kout] (row-major) into WMMA B-operand layout,
// f16. Layout per (kTile, nTile): 32 lanes x 16 halves. Lane n (0..15) = col,
// lanes 16..31 repeat cols with K offset +16. VGPR v packs K = 2v, 2v+1.
// dst dword index == flat thread index by construction.
// ---------------------------------------------------------------------------
__global__ __launch_bounds__(256) void pack_w_kernel(const float* __restrict__ W,
                                                     _Float16* __restrict__ Wp,
                                                     int Kin, int Kout,
                                                     int kTiles, int nTiles) {
    int idx = blockIdx.x * blockDim.x + threadIdx.x;
    int total = kTiles * nTiles * 32 * 8;
    if (idx >= total) return;
    int v    = idx & 7;
    int lane = (idx >> 3) & 31;
    int tile = idx >> 8;
    int nt = tile % nTiles;
    int kt = tile / nTiles;
    int khalf = (lane >> 4) * 16;
    int k0 = kt * 32 + khalf + 2 * v;
    int n  = nt * 16 + (lane & 15);
    float f0 = (k0     < Kin && n < Kout) ? W[(size_t)k0 * Kout + n]       : 0.0f;
    float f1 = (k0 + 1 < Kin && n < Kout) ? W[(size_t)(k0 + 1) * Kout + n] : 0.0f;
    union { _Float16 h[2]; unsigned u; } pk;
    pk.h[0] = (_Float16)f0;
    pk.h[1] = (_Float16)f1;
    ((unsigned*)Wp)[idx] = pk.u;
}

// ---------------------------------------------------------------------------
// Generic C = act(A @ W + b), f16 WMMA with f32 accumulate.
// A: [N, lda] fp32 row-major, effective K = Kin (zero padded to kTiles*32).
// Block = 256 threads = 8 waves (2 row-tiles x 4 col-tiles) over a 32x64 C
// block; grid = N/32. The 32x32 fp32 A tile is staged in LDS with
// GLOBAL_LOAD_ASYNC_TO_LDS_B128 (ASYNCcnt), double-buffered so K-step kt+1
// streams from HBM while WMMA consumes kt. fp32 -> f16 conversion happens at
// consume time (v_cvt_pk_f16_f32 co-issues with the matrix pipe).
// ---------------------------------------------------------------------------
__global__ __launch_bounds__(256) void gemm_relu_kernel(const float* __restrict__ A,
                                                        int lda, int Kin, int kTiles,
                                                        const _Float16* __restrict__ Wp,
                                                        int nTiles,
                                                        const float* __restrict__ bias,
                                                        float* __restrict__ C,
                                                        int ldc, int Kout, int doRelu) {
    // only shared object in this kernel; rows padded to 36 floats (144B, 16B-aligned)
    __shared__ alignas(16) float sA[2][32][36];

    const int tid     = threadIdx.x;
    const int lane    = tid & 31;
    const int wave    = tid >> 5;
    const int rowTile = wave >> 2;     // 0..1
    const int colTile = wave & 3;      // 0..3
    const int rowBase = blockIdx.x * 32;

    // staging role: each thread owns one 16B chunk of the 32x32 tile
    const int sr = tid >> 3;           // 0..31 (row)
    const int sc = (tid & 7) * 4;      // 0,4,...,28 (col)
    const float* gRow = A + (size_t)(rowBase + sr) * lda + sc;

    auto stage = [&](int buf, int kt) {
        const int kBase = kt * 32;
        const float* g = gRow + kBase;
        if (kBase + 32 <= Kin) {
#if USE_ASYNC
            unsigned loff = (unsigned)(uintptr_t)&sA[buf][sr][sc];
            asm volatile("global_load_async_to_lds_b128 %0, %1, off"
                         :: "v"(loff), "v"(g) : "memory");
#else
            *(float4*)&sA[buf][sr][sc] = *(const float4*)g;
#endif
        } else {  // K tail: bounds-checked scalar fill (zero pad)
            float4 v;
            v.x = (kBase + sc + 0 < Kin) ? g[0] : 0.0f;
            v.y = (kBase + sc + 1 < Kin) ? g[1] : 0.0f;
            v.z = (kBase + sc + 2 < Kin) ? g[2] : 0.0f;
            v.w = (kBase + sc + 3 < Kin) ? g[3] : 0.0f;
            *(float4*)&sA[buf][sr][sc] = v;
        }
    };

    v8f acc = {};
    int buf = 0;
    stage(0, 0);
#if USE_ASYNC
    asm volatile("s_wait_asynccnt 0x0" ::: "memory");
#endif
    __syncthreads();

    for (int kt = 0; kt < kTiles; ++kt) {
        if (kt + 1 < kTiles) stage(buf ^ 1, kt + 1);   // prefetch next K-step

        if (colTile < nTiles) {
            // A operand: lane l<16 -> row l, K {0..15}; lane>=16 -> K {16..31}
            int r  = rowTile * 16 + (lane & 15);
            int kb = (lane >> 4) * 16;
            const float* sp = &sA[buf][r][0];
            float4 f0 = *(const float4*)(sp + kb);
            float4 f1 = *(const float4*)(sp + kb + 4);
            float4 f2 = *(const float4*)(sp + kb + 8);
            float4 f3 = *(const float4*)(sp + kb + 12);
            union { _Float16 h[16]; v16h v; } au;
            au.h[0]  = (_Float16)f0.x; au.h[1]  = (_Float16)f0.y;
            au.h[2]  = (_Float16)f0.z; au.h[3]  = (_Float16)f0.w;
            au.h[4]  = (_Float16)f1.x; au.h[5]  = (_Float16)f1.y;
            au.h[6]  = (_Float16)f1.z; au.h[7]  = (_Float16)f1.w;
            au.h[8]  = (_Float16)f2.x; au.h[9]  = (_Float16)f2.y;
            au.h[10] = (_Float16)f2.z; au.h[11] = (_Float16)f2.w;
            au.h[12] = (_Float16)f3.x; au.h[13] = (_Float16)f3.y;
            au.h[14] = (_Float16)f3.z; au.h[15] = (_Float16)f3.w;

            v16h b = *(const v16h*)(Wp + (((size_t)kt * nTiles + colTile) * 32 + lane) * 16);
            acc = __builtin_amdgcn_wmma_f32_16x16x32_f16(false, au.v, false, b,
                                                         (short)0, acc, false, false);
        }
#if USE_ASYNC
        asm volatile("s_wait_asynccnt 0x0" ::: "memory");
#endif
        __syncthreads();
        buf ^= 1;
    }

    if (colTile < nTiles) {
        int colBase = colTile * 16 + (lane & 15);
        if (colBase < Kout) {
            float bv = bias[colBase];
            int rBase = rowBase + rowTile * 16 + ((lane >> 4) * 8);
            #pragma unroll
            for (int v = 0; v < 8; ++v) {
                float val = acc[v] + bv;
                if (doRelu) val = fmaxf(val, 0.0f);
                C[(size_t)(rBase + v) * ldc + colBase] = val;
            }
        }
    }
}

// ---------------------------------------------------------------------------
// Degree: deg[src[e]] += 1.0 (float atomics -> global_atomic_add_f32)
// ---------------------------------------------------------------------------
__global__ __launch_bounds__(256) void deg_kernel(const int* __restrict__ src,
                                                  float* __restrict__ deg, int E) {
    int e = blockIdx.x * blockDim.x + threadIdx.x;
    if (e >= E) return;
    __hip_atomic_fetch_add(&deg[src[e]], 1.0f, __ATOMIC_RELAXED, __HIP_MEMORY_SCOPE_AGENT);
}

__global__ __launch_bounds__(256) void invdeg_kernel(float* __restrict__ deg, int n) {
    int i = blockIdx.x * blockDim.x + threadIdx.x;
    if (i >= n) return;
    deg[i] = 1.0f / fmaxf(deg[i], 1.0f);
}

// ---------------------------------------------------------------------------
// SpMM scatter: one wave per edge; lane handles 2 of the 64 feature dims.
// out[src] += h[dst] * inv_deg[src]
// ---------------------------------------------------------------------------
__global__ __launch_bounds__(256) void spmm_kernel(const int* __restrict__ src,
                                                   const int* __restrict__ dst,
                                                   const float* __restrict__ inv_deg,
                                                   const float* __restrict__ h,
                                                   float* __restrict__ out, int E) {
    int e = blockIdx.x * (blockDim.x >> 5) + (threadIdx.x >> 5);
    if (e >= E) return;
    int lane = threadIdx.x & 31;
    int s = src[e];
    int d = dst[e];
    float w = inv_deg[s];
    float2 v = *(const float2*)(h + (size_t)d * HID + lane * 2);
    float* op = out + (size_t)s * HID + lane * 2;
    __hip_atomic_fetch_add(op + 0, v.x * w, __ATOMIC_RELAXED, __HIP_MEMORY_SCOPE_AGENT);
    __hip_atomic_fetch_add(op + 1, v.y * w, __ATOMIC_RELAXED, __HIP_MEMORY_SCOPE_AGENT);
}

// ---------------------------------------------------------------------------
// Copy a [N,64] fp32 block into dst[:, colOff:colOff+64] with row stride ldDst.
// ---------------------------------------------------------------------------
__global__ __launch_bounds__(256) void copy_block_kernel(const float* __restrict__ src,
                                                         float* __restrict__ dst,
                                                         int ldDst, int colOff, int n) {
    int idx = blockIdx.x * blockDim.x + threadIdx.x;     // over n*16 float4s
    if (idx >= n * 16) return;
    int row = idx >> 4;
    int c4  = idx & 15;
    float4 v = *(const float4*)(src + (size_t)row * HID + c4 * 4);
    *(float4*)(dst + (size_t)row * ldDst + colOff + c4 * 4) = v;
}

// ---------------------------------------------------------------------------
extern "C" void kernel_launch(void* const* d_in, const int* in_sizes, int n_in,
                              void* d_out, int out_size, void* d_ws, size_t ws_size,
                              hipStream_t stream) {
    const float* x       = (const float*)d_in[0];
    const int*   e1      = (const int*)d_in[1];
    const int*   e2      = (const int*)d_in[2];
    const float* W_embed = (const float*)d_in[3];
    const float* b_embed = (const float*)d_in[4];
    const float* W0      = (const float*)d_in[5];
    const float* b0      = (const float*)d_in[6];
    const float* W1      = (const float*)d_in[7];
    const float* b1      = (const float*)d_in[8];
    const float* Wc      = (const float*)d_in[9];
    const float* bc      = (const float*)d_in[10];
    float* logits = (float*)d_out;

    const int N = in_sizes[0] / IN_DIM;   // 100000
    const int E = in_sizes[1] / 2;        // 1200000

    // ---- workspace carve-up ----
    char* ws = (char*)d_ws;
    size_t off = 0;
    auto carve = [&](size_t bytes) -> void* {
        void* p = ws + off;
        off += (bytes + 255) & ~(size_t)255;
        return p;
    };
    float* r0   = (float*)carve((size_t)N * HID * 4);
    float* r1   = (float*)carve((size_t)N * HID * 4);
    float* r2   = (float*)carve((size_t)N * HID * 4);
    float* nb1  = (float*)carve((size_t)N * HID * 4);
    float* nb2  = (float*)carve((size_t)N * HID * 4);
    float* cat  = (float*)carve((size_t)N * 256 * 4);   // also reused for emb [N,192]
    float* inv1 = (float*)carve((size_t)N * 4);
    float* inv2 = (float*)carve((size_t)N * 4);
    const int ktE = (IN_DIM + 31) / 32;   // 16
    const int kt0 = 256 / 32;             // 8
    const int ktC = 192 / 32;             // 6
    const int ntH = HID / 16;             // 4
    const int ntC = (NCLS + 15) / 16;     // 3
    _Float16* pWe = (_Float16*)carve((size_t)ktE * ntH * 512 * 2);
    _Float16* pW0 = (_Float16*)carve((size_t)kt0 * ntH * 512 * 2);
    _Float16* pW1 = (_Float16*)carve((size_t)kt0 * ntH * 512 * 2);
    _Float16* pWc = (_Float16*)carve((size_t)ktC * ntC * 512 * 2);

    const int TB = 256;
    const int gemmBlocks = N / 32;        // 3125
    const int spmmBlocks = (E + 7) / 8;   // 8 waves / block

    // ---- pack weights into WMMA B-operand layout ----
    {
        int t;
        t = ktE * ntH * 256;
        hipLaunchKernelGGL(pack_w_kernel, dim3((t + TB - 1) / TB), dim3(TB), 0, stream,
                           W_embed, pWe, IN_DIM, HID, ktE, ntH);
        t = kt0 * ntH * 256;
        hipLaunchKernelGGL(pack_w_kernel, dim3((t + TB - 1) / TB), dim3(TB), 0, stream,
                           W0, pW0, 256, HID, kt0, ntH);
        hipLaunchKernelGGL(pack_w_kernel, dim3((t + TB - 1) / TB), dim3(TB), 0, stream,
                           W1, pW1, 256, HID, kt0, ntH);
        t = ktC * ntC * 256;
        hipLaunchKernelGGL(pack_w_kernel, dim3((t + TB - 1) / TB), dim3(TB), 0, stream,
                           Wc, pWc, 192, NCLS, ktC, ntC);
    }

    // ---- inverse degrees (edge-only, shared by both layers) ----
    hipMemsetAsync(inv1, 0, (size_t)N * 4, stream);
    hipMemsetAsync(inv2, 0, (size_t)N * 4, stream);
    hipLaunchKernelGGL(deg_kernel, dim3((E + TB - 1) / TB), dim3(TB), 0, stream, e1, inv1, E);
    hipLaunchKernelGGL(deg_kernel, dim3((E + TB - 1) / TB), dim3(TB), 0, stream, e2, inv2, E);
    hipLaunchKernelGGL(invdeg_kernel, dim3((N + TB - 1) / TB), dim3(TB), 0, stream, inv1, N);
    hipLaunchKernelGGL(invdeg_kernel, dim3((N + TB - 1) / TB), dim3(TB), 0, stream, inv2, N);

    // ---- r0 = relu(x @ W_embed + b_embed) ----
    hipLaunchKernelGGL(gemm_relu_kernel, dim3(gemmBlocks), dim3(TB), 0, stream,
                       x, IN_DIM, IN_DIM, ktE, pWe, ntH, b_embed, r0, HID, HID, 1);

    // ---- 2 H2GCN layers ----
    float* rPrev = r0;
    float* rOuts[2] = {r1, r2};
    const _Float16* pWl[2] = {pW0, pW1};
    const float* bl[2] = {b0, b1};
    for (int l = 0; l < 2; ++l) {
        hipMemsetAsync(nb1, 0, (size_t)N * HID * 4, stream);
        hipMemsetAsync(nb2, 0, (size_t)N * HID * 4, stream);
        hipLaunchKernelGGL(spmm_kernel, dim3(spmmBlocks), dim3(TB), 0, stream,
                           e1, e1 + E, inv1, rPrev, nb1, E);
        hipLaunchKernelGGL(spmm_kernel, dim3(spmmBlocks), dim3(TB), 0, stream,
                           e2, e2 + E, inv2, rPrev, nb2, E);
        // cat = [rPrev | nb1 | rPrev | nb2]  (ego separation)
        int cb = (N * 16 + TB - 1) / TB;
        hipLaunchKernelGGL(copy_block_kernel, dim3(cb), dim3(TB), 0, stream, rPrev, cat, 256, 0,   N);
        hipLaunchKernelGGL(copy_block_kernel, dim3(cb), dim3(TB), 0, stream, nb1,   cat, 256, 64,  N);
        hipLaunchKernelGGL(copy_block_kernel, dim3(cb), dim3(TB), 0, stream, rPrev, cat, 256, 128, N);
        hipLaunchKernelGGL(copy_block_kernel, dim3(cb), dim3(TB), 0, stream, nb2,   cat, 256, 192, N);
        // r_next = relu(cat @ Wl + bl)
        hipLaunchKernelGGL(gemm_relu_kernel, dim3(gemmBlocks), dim3(TB), 0, stream,
                           cat, 256, 256, kt0, pWl[l], ntH, bl[l], rOuts[l], HID, HID, 1);
        rPrev = rOuts[l];
    }

    // ---- emb = [r0 | r1 | r2] (reuse cat buffer, ld = 192) ----
    {
        int cb = (N * 16 + TB - 1) / TB;
        hipLaunchKernelGGL(copy_block_kernel, dim3(cb), dim3(TB), 0, stream, r0, cat, 192, 0,   N);
        hipLaunchKernelGGL(copy_block_kernel, dim3(cb), dim3(TB), 0, stream, r1, cat, 192, 64,  N);
        hipLaunchKernelGGL(copy_block_kernel, dim3(cb), dim3(TB), 0, stream, r2, cat, 192, 128, N);
    }

    // ---- logits = emb @ Wc + bc (no relu, Kout=47 masked) ----
    hipLaunchKernelGGL(gemm_relu_kernel, dim3(gemmBlocks), dim3(TB), 0, stream,
                       cat, 192, 192, ktC, pWc, ntC, bc, logits, NCLS, NCLS, 0);
}
